// MambaModel_68942815035942
// MI455X (gfx1250) — matrix operand order repaired and checked by reading, not compiled, not run on hardware
//
#include <hip/hip_runtime.h>
#include <hip/hip_bf16.h>

// ---------------------------------------------------------------------------
// Mamba forward for MI455X (gfx1250), wave32 + WMMA bf16 tensor path + TDM.
// Dims fixed by the reference: B=4, L=1024, D=1024, E=2048, N=16, K=4, R=64.
// ---------------------------------------------------------------------------

static constexpr int Bsz  = 4;
static constexpr int Lseq = 1024;
static constexpr int Dm   = 1024;
static constexpr int Em   = 2 * Dm;      // 2048
static constexpr int Nst  = 16;
static constexpr int Kc   = 4;
static constexpr int Rr   = Dm / 16;     // 64
static constexpr int NLAY = 4;
static constexpr float LN_EPS = 1e-5f;

typedef __attribute__((ext_vector_type(16))) __bf16 v16bf;
typedef __attribute__((ext_vector_type(8)))  __bf16 v8bf;
typedef __attribute__((ext_vector_type(8)))  float  v8f;
typedef __attribute__((ext_vector_type(4)))  float  v4f;
typedef __attribute__((ext_vector_type(4)))  unsigned int v4u;
typedef __attribute__((ext_vector_type(4)))  unsigned int u32x4;
typedef __attribute__((ext_vector_type(4)))  int  i32x4;
typedef __attribute__((ext_vector_type(8)))  int  i32x8;

#if defined(__has_builtin)
#if __has_builtin(__builtin_amdgcn_tensor_load_to_lds) && \
    __has_builtin(__builtin_amdgcn_s_wait_tensorcnt)
#define HAVE_TDM 1
#endif
#endif

__device__ __forceinline__ unsigned short f2bf(float f) {
    union { float f; unsigned u; } v; v.f = f;
    unsigned r = v.u + 0x7FFFu + ((v.u >> 16) & 1u);   // round-to-nearest-even
    return (unsigned short)(r >> 16);
}
__device__ __forceinline__ unsigned pack2(float a, float b) {
    return (unsigned)f2bf(a) | ((unsigned)f2bf(b) << 16);
}

#ifdef HAVE_TDM
// D# group 0 (ISA 8.3): count=1 | lds_addr | global_addr[56:0] | type=2
__device__ __forceinline__ u32x4 tdm_g0(unsigned ldsOff, const void* gp) {
    unsigned long long ga = (unsigned long long)gp;
    u32x4 g0;
    g0.x = 1u;
    g0.y = ldsOff;
    g0.z = (unsigned)(ga & 0xFFFFFFFFull);
    g0.w = (unsigned)((ga >> 32) & 0x01FFFFFFull) | (2u << 30);
    return g0;
}
#if defined(__clang_major__) && __clang_major__ >= 23
#define TDM_LOAD(g0, g1, gz4, gz8) \
    __builtin_amdgcn_tensor_load_to_lds((g0), (g1), (gz4), (gz4), (gz8), 0)
#else
#define TDM_LOAD(g0, g1, gz4, gz8) \
    __builtin_amdgcn_tensor_load_to_lds((g0), (g1), (gz4), (gz4), 0)
#endif
#endif  // HAVE_TDM

// ---------------------------------------------------------------------------
// (B,L,D) -> (B,D,L) transpose, LDS tiled.
// ---------------------------------------------------------------------------
__global__ void k_transpose_bdl(const float* __restrict__ x, float* __restrict__ h) {
    __shared__ float tile[32][33];
    const int b = blockIdx.z;
    const int l0 = blockIdx.x * 32, d0 = blockIdx.y * 32;
    const int tx = threadIdx.x, ty = threadIdx.y;        // (32, 8)
    #pragma unroll
    for (int i = 0; i < 4; ++i)
        tile[ty + i * 8][tx] = x[((size_t)b * Lseq + (l0 + ty + i * 8)) * Dm + d0 + tx];
    __syncthreads();
    #pragma unroll
    for (int i = 0; i < 4; ++i)
        h[((size_t)b * Dm + (d0 + ty + i * 8)) * Lseq + l0 + tx] = tile[tx][ty + i * 8];
}

// ---------------------------------------------------------------------------
// C[M,N] = A[M,K] * B[K,N], row-major fp32 in / fp32 out, bf16 WMMA inside.
// Block tile 128x64, K-step 64.  8 waves = 4(M) x 2(N); each wave owns a
// 32x32 output (2x2 tiles of 16x16) -> 8 v_wmma per LDS stage, with
// A-fragment reuse across N tiles and B-fragment reuse across M tiles.
// K must be a multiple of 64 (call sites: 1024, 2048, 64).
// blockIdx.z = batch (B operand / C strided, A shared).
// ---------------------------------------------------------------------------
__global__ __launch_bounds__(256)
void k_gemm_wmma_bf16(const float* __restrict__ A, const float* __restrict__ Bg,
                      float* __restrict__ C, int M, int N, int K,
                      int lda, int ldb, int ldc,
                      long long strideB, long long strideC)
{
    __shared__ __align__(16) unsigned short lA[128][72]; // 128x64 bf16, 144B stride
    __shared__ __align__(16) unsigned short lB[64][64];  // transposed: [n][k]

    const int tid  = threadIdx.x;
    const int wave = tid >> 5, lane = tid & 31;
    const int half = lane >> 4, lr = lane & 15;
    const int wm = wave >> 1, wn = wave & 1;             // 4x2 wave grid
    const int mBlk = blockIdx.y * 128;
    const int nBlk = blockIdx.x * 64;

    const float* Bb = Bg + (size_t)blockIdx.z * strideB;
    float*       Cb = C  + (size_t)blockIdx.z * strideC;

    v8f acc[2][2] = {};

    // staging roles
    const int arow = tid >> 1, acol = (tid & 1) * 32;    // A: 2 thr/row, 32 flt each
    const int bn = tid & 63,  bkq = (tid >> 6) * 16;     // B: 16 flt/thr, k-strided

    for (int k0 = 0; k0 < K; k0 += 64) {
        // ---- stage A tile (fp32 -> bf16), 8x float4 per thread ----
        {
            const int gm = mBlk + arow;
            const float* p = A + (size_t)(gm < M ? gm : 0) * lda + k0 + acol;
            #pragma unroll
            for (int j = 0; j < 4; ++j) {
                v4f f0 = {}, f1 = {};
                if (gm < M) { f0 = *(const v4f*)(p + j * 8); f1 = *(const v4f*)(p + j * 8 + 4); }
                v4u s;
                s.x = pack2(f0.x, f0.y); s.y = pack2(f0.z, f0.w);
                s.z = pack2(f1.x, f1.y); s.w = pack2(f1.z, f1.w);
                *(v4u*)&lA[arow][acol + j * 8] = s;
            }
            __builtin_prefetch(p + 64, 0, 1);            // global_prefetch_b8 next K tile
        }
        // ---- stage B tile transposed into lB[n][k] ----
        {
            const float* p = Bb + (size_t)(k0 + bkq) * ldb + nBlk + bn;
            #pragma unroll
            for (int j = 0; j < 2; ++j) {
                float f[8];
                #pragma unroll
                for (int q = 0; q < 8; ++q) f[q] = p[(size_t)(j * 8 + q) * ldb];
                v4u s;
                s.x = pack2(f[0], f[1]); s.y = pack2(f[2], f[3]);
                s.z = pack2(f[4], f[5]); s.w = pack2(f[6], f[7]);
                *(v4u*)&lB[bn][bkq + j * 8] = s;
            }
        }
        __syncthreads();

        #pragma unroll
        for (int ks = 0; ks < 64; ks += 32) {
            // A fragments (ISA 7.12.2 16-bit A 16x32 layout; 2x 16B LDS loads/lane)
            v16bf afr[2];
            #pragma unroll
            for (int mi = 0; mi < 2; ++mi) {
                const unsigned short* ap = &lA[wm * 32 + mi * 16 + lr][ks];
                v8bf a0 = *(const v8bf*)(ap + half * 8);
                v8bf a1 = *(const v8bf*)(ap + 16 + half * 8);
                afr[mi] = __builtin_shufflevector(a0, a1,
                          0,1,2,3,4,5,6,7,8,9,10,11,12,13,14,15);
            }
            // B fragments (32x16; same k order as A, lB is [n][k])
            v16bf bfr[2];
            #pragma unroll
            for (int ni = 0; ni < 2; ++ni) {
                const unsigned short* bp = &lB[wn * 32 + ni * 16 + lr][ks + half * 8];
                v8bf b0 = *(const v8bf*)(bp);
                v8bf b1 = *(const v8bf*)(bp + 16);
                bfr[ni] = __builtin_shufflevector(b0, b1,
                          0,1,2,3,4,5,6,7,8,9,10,11,12,13,14,15);
            }
            #pragma unroll
            for (int mi = 0; mi < 2; ++mi)
                #pragma unroll
                for (int ni = 0; ni < 2; ++ni)
                    acc[mi][ni] = __builtin_amdgcn_wmma_f32_16x16x32_bf16(
                        false, afr[mi], false, bfr[ni], (short)0, acc[mi][ni],
                        false, false);
        }
        __syncthreads();
    }

    // ---- epilogue: C/D layout VGPR v -> M = v + 8*half, N = lane&15 ----
    #pragma unroll
    for (int mi = 0; mi < 2; ++mi) {
        #pragma unroll
        for (int ni = 0; ni < 2; ++ni) {
            const int gn = nBlk + wn * 32 + ni * 16 + lr;
            #pragma unroll
            for (int v = 0; v < 8; ++v) {
                const int gm = mBlk + wm * 32 + mi * 16 + half * 8 + v;
                if (gm < M && gn < N) Cb[(size_t)gm * ldc + gn] = acc[mi][ni][v];
            }
        }
    }
}

// ---------------------------------------------------------------------------
// Depthwise causal conv (K=4) + bias + SiLU.  u[b,e,l] from xz rows [0,E).
// ---------------------------------------------------------------------------
__global__ void k_conv_silu(const float* __restrict__ xz, const float* __restrict__ cw,
                            const float* __restrict__ cb, float* __restrict__ u)
{
    const size_t idx = (size_t)blockIdx.x * blockDim.x + threadIdx.x;   // B*E*L
    const int l = (int)(idx % Lseq);
    const size_t be = idx / Lseq;
    const int e = (int)(be % Em);
    const int b = (int)(be / Em);
    const float* row = xz + ((size_t)b * 2 * Em + e) * Lseq;
    const float* w = cw + (size_t)e * Kc;
    float s = cb[e];
    #pragma unroll
    for (int j = 0; j < Kc; ++j) {
        const int ll = l + j - (Kc - 1);
        if (ll >= 0) s += w[j] * row[ll];
    }
    u[((size_t)b * Em + e) * Lseq + l] = s / (1.0f + __expf(-s));
}

// ---------------------------------------------------------------------------
// delta = softplus(delta + bdt[e]) in place.
// ---------------------------------------------------------------------------
__global__ void k_softplus_bias(float* __restrict__ delta, const float* __restrict__ bdt)
{
    const size_t idx = (size_t)blockIdx.x * blockDim.x + threadIdx.x;   // B*E*L
    const int e = (int)((idx / Lseq) % Em);
    const float v = delta[idx] + bdt[e];
    delta[idx] = (v > 20.0f) ? v : __logf(1.0f + __expf(v));
}

// ---------------------------------------------------------------------------
// Selective scan: thread = one channel e, N=16 states in registers.  The
// shared B/C chunks (16 x 64 fp32, row stride L) are staged to LDS by the
// Tensor Data Mover (wave 0 issues, s_wait_tensorcnt, workgroup barrier);
// falls back to cooperative VALU staging when the TDM builtin is absent.
// Fuses +Dp*u and *silu(zgate).
// ---------------------------------------------------------------------------
__global__ __launch_bounds__(256)
void k_scan(const float* __restrict__ delta, const float* __restrict__ u,
            const float* __restrict__ xdbl,  const float* __restrict__ xz,
            const float* __restrict__ Alog,  const float* __restrict__ Dp,
            float* __restrict__ y)
{
    const int tid = threadIdx.x;
    const int b = blockIdx.y;
    const int e = blockIdx.x * 256 + tid;

    const float* dre = delta + ((size_t)b * Em + e) * Lseq;
    const float* ure = u     + ((size_t)b * Em + e) * Lseq;
    const float* zre = xz    + ((size_t)b * 2 * Em + Em + e) * Lseq;   // zgate
    const float* Bm  = xdbl + (size_t)b * (Rr + 2 * Nst) * Lseq + (size_t)Rr * Lseq;
    const float* Cm  = Bm + (size_t)Nst * Lseq;
    float*       yre = y + ((size_t)b * Em + e) * Lseq;

    float Aen[Nst], hs[Nst];
    #pragma unroll
    for (int n = 0; n < Nst; ++n) {
        Aen[n] = -__expf(Alog[(size_t)e * Nst + n]);
        hs[n] = 0.0f;
    }
    const float dp = Dp[e];

    __shared__ __align__(16) float sB[Nst][64];
    __shared__ __align__(16) float sC[Nst][64];

#ifdef HAVE_TDM
    // D# group 1 (ISA 8.4): data_size=4B; tensor_dim0=L; tensor_dim1=16;
    // tile 64x16; tensor_dim0_stride=L.  Shared by the B and C descriptors.
    i32x8 g1;
    g1[0] = 0x20000;                       // workgroup_mask=0, data_size=2 (4B)
    g1[1] = (int)((unsigned)Lseq << 16);   // tensor_dim0[15:0] << 16
    g1[2] = (int)(16u << 16);              // tensor_dim0 hi=0 | tensor_dim1=16
    g1[3] = (int)(64u << 16);              // tensor_dim1 hi=0 | tile_dim0=64
    g1[4] = 16;                            // tile_dim1=16, tile_dim2=0
    g1[5] = Lseq;                          // tensor_dim0_stride (48b, low 32)
    g1[6] = 0;
    g1[7] = 0;
    const i32x4 gz4 = {};
    const i32x8 gz8 = {};
    (void)gz8;
    const unsigned ldsB = (unsigned)(size_t)&sB[0][0];
    const unsigned ldsC = (unsigned)(size_t)&sC[0][0];
#endif

    for (int l0 = 0; l0 < Lseq; l0 += 64) {
        __syncthreads();                   // prior chunk fully consumed
#ifdef HAVE_TDM
        if (tid < 32) {                    // wave 0 drives the TDM
            TDM_LOAD(tdm_g0(ldsB, (const void*)(Bm + l0)), g1, gz4, gz8);
            TDM_LOAD(tdm_g0(ldsC, (const void*)(Cm + l0)), g1, gz4, gz8);
            __builtin_amdgcn_s_wait_tensorcnt(0);
        }
#else
        {   // cooperative stage: 16x64 floats each for B and C, 4 per thread
            const int n = tid >> 4, lo = (tid & 15) * 4;
            *(v4f*)&sB[n][lo] = *(const v4f*)(Bm + (size_t)n * Lseq + l0 + lo);
            *(v4f*)&sC[n][lo] = *(const v4f*)(Cm + (size_t)n * Lseq + l0 + lo);
        }
#endif
        __syncthreads();
        for (int t = 0; t < 64; ++t) {
            const int l = l0 + t;
            const float d = dre[l], uu = ure[l], z = zre[l];
            const float du = d * uu;
            float accv = 0.0f;
            #pragma unroll
            for (int n = 0; n < Nst; ++n) {
                const float hn = __expf(d * Aen[n]) * hs[n] + du * sB[n][t];
                hs[n] = hn;
                accv += hn * sC[n][t];
            }
            const float yy = accv + dp * uu;
            yre[l] = yy * (z / (1.0f + __expf(-z)));
        }
    }
}

// ---------------------------------------------------------------------------
// h = LayerNorm_D(h + z) in place; one block per (l, b).
// ---------------------------------------------------------------------------
__global__ __launch_bounds__(256)
void k_ln_residual(float* __restrict__ h, const float* __restrict__ z,
                   const float* __restrict__ g, const float* __restrict__ beta)
{
    const int l = blockIdx.x, b = blockIdx.y, tid = threadIdx.x;
    __shared__ float red[256];
    float v[4];
    float s = 0.0f;
    #pragma unroll
    for (int i = 0; i < 4; ++i) {
        const int d = tid + i * 256;
        const size_t idx = ((size_t)b * Dm + d) * Lseq + l;
        v[i] = h[idx] + z[idx];
        s += v[i];
    }
    red[tid] = s; __syncthreads();
    for (int o = 128; o > 0; o >>= 1) { if (tid < o) red[tid] += red[tid + o]; __syncthreads(); }
    const float mean = red[0] * (1.0f / Dm);
    __syncthreads();
    float sq = 0.0f;
    #pragma unroll
    for (int i = 0; i < 4; ++i) { const float t = v[i] - mean; sq += t * t; }
    red[tid] = sq; __syncthreads();
    for (int o = 128; o > 0; o >>= 1) { if (tid < o) red[tid] += red[tid + o]; __syncthreads(); }
    const float inv = rsqrtf(red[0] * (1.0f / Dm) + LN_EPS);
    #pragma unroll
    for (int i = 0; i < 4; ++i) {
        const int d = tid + i * 256;
        const size_t idx = ((size_t)b * Dm + d) * Lseq + l;
        h[idx] = (v[i] - mean) * inv * g[d] + beta[d];
    }
}

// ---------------------------------------------------------------------------
// out[b,d] = mean_l h[b,d,l]
// ---------------------------------------------------------------------------
__global__ void k_mean_l(const float* __restrict__ h, float* __restrict__ out)
{
    const int d = blockIdx.x, b = blockIdx.y, tid = threadIdx.x;
    const float* row = h + ((size_t)b * Dm + d) * Lseq;
    float s = 0.0f;
    for (int l = tid; l < Lseq; l += 256) s += row[l];
    __shared__ float red[256];
    red[tid] = s; __syncthreads();
    for (int o = 128; o > 0; o >>= 1) { if (tid < o) red[tid] += red[tid + o]; __syncthreads(); }
    if (tid == 0) out[(size_t)b * Dm + d] = red[0] * (1.0f / Lseq);
}

// ---------------------------------------------------------------------------
extern "C" void kernel_launch(void* const* d_in, const int* in_sizes, int n_in,
                              void* d_out, int out_size, void* d_ws, size_t ws_size,
                              hipStream_t stream)
{
    (void)in_sizes; (void)n_in; (void)out_size; (void)ws_size;

    const float* x     = (const float*)d_in[0];
    const float* Win   = (const float*)d_in[1];
    const float* convw = (const float*)d_in[2];
    const float* convb = (const float*)d_in[3];
    const float* Wx    = (const float*)d_in[4];
    const float* Wdt   = (const float*)d_in[5];
    const float* bdt   = (const float*)d_in[6];
    const float* Alog  = (const float*)d_in[7];
    const float* Dp    = (const float*)d_in[8];
    const float* Wout  = (const float*)d_in[9];
    const float* ln_g  = (const float*)d_in[10];
    const float* ln_b  = (const float*)d_in[11];
    float* out = (float*)d_out;

    const int Rd = Rr + 2 * Nst;   // 96 rows of x_dbl

    float* ws = (float*)d_ws;
    size_t off = 0;
    float* h    = ws + off; off += (size_t)Bsz * Dm * Lseq;       // 16 MB
    float* xzb  = ws + off; off += (size_t)Bsz * 2 * Em * Lseq;   // 64 MB
    float* ub   = ws + off; off += (size_t)Bsz * Em * Lseq;       // 32 MB
    float* xdbl = ws + off; off += (size_t)Bsz * Rd * Lseq;       // 1.5 MB
    float* dlt  = ws + off; off += (size_t)Bsz * Em * Lseq;       // 32 MB
    float* yb   = ws + off; off += (size_t)Bsz * Em * Lseq;       // 32 MB
    float* zb   = ws + off; off += (size_t)Bsz * Dm * Lseq;       // 16 MB

    // h = swapaxes(x, -1, -2)
    k_transpose_bdl<<<dim3(Lseq / 32, Dm / 32, Bsz), dim3(32, 8), 0, stream>>>(x, h);

    const size_t tot = (size_t)Bsz * Em * Lseq;

    for (int i = 0; i < NLAY; ++i) {
        const float* Win_i   = Win   + (size_t)i * 2 * Em * Dm;
        const float* convw_i = convw + (size_t)i * Em * Kc;
        const float* convb_i = convb + (size_t)i * Em;
        const float* Wx_i    = Wx    + (size_t)i * Rd * Em;
        const float* Wdt_i   = Wdt   + (size_t)i * Em * Rr;
        const float* bdt_i   = bdt   + (size_t)i * Em;
        const float* Alog_i  = Alog  + (size_t)i * Em * Nst;
        const float* Dp_i    = Dp    + (size_t)i * Em;
        const float* Wout_i  = Wout  + (size_t)i * Dm * Em;
        const float* lng_i   = ln_g  + (size_t)i * Dm;
        const float* lnb_i   = ln_b  + (size_t)i * Dm;

        // 1) xz[b] = Win_i (2E x D) @ h[b] (D x L)
        k_gemm_wmma_bf16<<<dim3(Lseq / 64, (2 * Em + 127) / 128, Bsz), 256, 0, stream>>>(
            Win_i, h, xzb, 2 * Em, Lseq, Dm, Dm, Lseq, Lseq,
            (long long)Dm * Lseq, (long long)2 * Em * Lseq);

        // 2) u = silu(causal_conv(xz[:, :E]) + convb)
        k_conv_silu<<<dim3((unsigned)(tot / 256)), 256, 0, stream>>>(xzb, convw_i, convb_i, ub);

        // 3) x_dbl[b] = Wx_i (96 x E) @ u[b] (E x L)
        k_gemm_wmma_bf16<<<dim3(Lseq / 64, (Rd + 127) / 128, Bsz), 256, 0, stream>>>(
            Wx_i, ub, xdbl, Rd, Lseq, Em, Em, Lseq, Lseq,
            (long long)Em * Lseq, (long long)Rd * Lseq);

        // 4) delta = softplus(Wdt_i (E x R) @ dtr[b] (R x L) + bdt)
        k_gemm_wmma_bf16<<<dim3(Lseq / 64, Em / 128, Bsz), 256, 0, stream>>>(
            Wdt_i, xdbl, dlt, Em, Lseq, Rr, Rr, Lseq, Lseq,
            (long long)Rd * Lseq, (long long)Em * Lseq);
        k_softplus_bias<<<dim3((unsigned)(tot / 256)), 256, 0, stream>>>(dlt, bdt_i);

        // 5) selective scan (fused +Dp*u and *silu(zgate))
        k_scan<<<dim3(Em / 256, Bsz), 256, 0, stream>>>(dlt, ub, xdbl, xzb, Alog_i, Dp_i, yb);

        // 6) z[b] = Wout_i (D x E) @ y[b] (E x L)
        k_gemm_wmma_bf16<<<dim3(Lseq / 64, Dm / 128, Bsz), 256, 0, stream>>>(
            Wout_i, yb, zb, Dm, Lseq, Em, Em, Lseq, Lseq,
            (long long)Em * Lseq, (long long)Dm * Lseq);

        // 7) h = LN(h + z)
        k_ln_residual<<<dim3(Lseq, Bsz), 256, 0, stream>>>(h, zb, lng_i, lnb_i);
    }

    // out = mean over L
    k_mean_l<<<dim3(Dm, Bsz), 256, 0, stream>>>(h, out);
}
